// YatConv_19791209300379
// MI455X (gfx1250) — matrix operand order repaired and checked by reading, not compiled, not run on hardware
//
#include <hip/hip_runtime.h>

// ---------------------------------------------------------------------------
// YatConv for MI455X (gfx1250): implicit-GEMM 3x3 conv via v_wmma_f32_16x16x32_bf16
// with fp32 = bf16_hi + bf16_lo error-compensated split (3 WMMAs per tile-step).
// x and W are pre-split to bf16 hi/lo planes in workspace; the hot loop is a
// 2-deep software pipeline: CDNA5 async global->LDS copies (ASYNCcnt, in-order
// retirement -> s_wait_asynccnt 8) into ping-pong LDS buffers overlapped with
// WMMA on the other buffer.
// Dims fixed per reference: x[16,64,64,128] f32, W[3,3,128,256] f32 (HWIO),
// bias[256], alpha[1].  out[16,64,64,256] f32.
// ---------------------------------------------------------------------------

typedef __attribute__((ext_vector_type(16))) __bf16 bf16x16;
typedef __attribute__((ext_vector_type(8)))  float  f32x8;

#define YC_EPS 1e-5f

// ---- bf16 helpers ----------------------------------------------------------
__device__ __forceinline__ unsigned short f32_to_bf16_rne(float f) {
    unsigned int u = __builtin_bit_cast(unsigned int, f);
    u += 0x7FFFu + ((u >> 16) & 1u);                 // round to nearest even
    return (unsigned short)(u >> 16);
}
__device__ __forceinline__ float bf16_bits_to_f32(unsigned short h) {
    unsigned int u = (unsigned int)h << 16;
    return __builtin_bit_cast(float, u);
}

struct Bytes32 { uint4 a, b; };
__device__ __forceinline__ bf16x16 make_frag(uint4 a, uint4 b) {
    Bytes32 t; t.a = a; t.b = b;
    return __builtin_bit_cast(bf16x16, t);
}

// ---- CDNA5 async global->LDS copy (ASYNCcnt path) --------------------------
// LDS target is the VDST byte offset plus the imm offset; global addr gets same offset.
template <int OFF>
__device__ __forceinline__ void async_b128(unsigned lds_byte_off, const void* gaddr) {
    asm volatile("global_load_async_to_lds_b128 %0, %1, off offset:%2"
                 :: "v"(lds_byte_off), "v"(gaddr), "n"(OFF)
                 : "memory");
}
template <int N>
__device__ __forceinline__ void wait_async() {
    asm volatile("s_wait_asynccnt %0" :: "n"(N) : "memory");
}
__device__ __forceinline__ unsigned lds_off(const void* p) {
    // generic LDS pointer = {SHARED_BASE aperture, 32-bit LDS offset}
    return (unsigned)(unsigned long long)p;
}

// ---------------------------------------------------------------------------
// Prep kernel 1: per-pixel sum of squares over 128 channels.  s[65536]
// ---------------------------------------------------------------------------
__global__ void yc_pix_sumsq(const float* __restrict__ x, float* __restrict__ s) {
    const int p = blockIdx.x * blockDim.x + threadIdx.x;   // 0..65535
    const float4* xp = (const float4*)(x + (size_t)p * 128);
    float acc = 0.0f;
#pragma unroll 8
    for (int i = 0; i < 32; ++i) {
        float4 v = xp[i];
        acc += v.x * v.x + v.y * v.y + v.z * v.z + v.w * v.w;
    }
    s[p] = acc;
}

// ---------------------------------------------------------------------------
// Prep kernel 2: 3x3 zero-padded box sum of s -> ps (patch squared norm)
// ---------------------------------------------------------------------------
__global__ void yc_boxsum(const float* __restrict__ s, float* __restrict__ ps) {
    const int p = blockIdx.x * blockDim.x + threadIdx.x;
    const int b = p >> 12, h = (p >> 6) & 63, w = p & 63;
    float acc = 0.0f;
#pragma unroll
    for (int dh = -1; dh <= 1; ++dh) {
#pragma unroll
        for (int dw = -1; dw <= 1; ++dw) {
            const int hh = h + dh, ww = w + dw;
            if ((unsigned)hh < 64u && (unsigned)ww < 64u)
                acc += s[(b << 12) + (hh << 6) + ww];
        }
    }
    ps[p] = acc;
}

// ---------------------------------------------------------------------------
// Prep kernel 3: split x fp32 -> bf16 hi/lo planes (same [p][c] layout)
// ---------------------------------------------------------------------------
__global__ void yc_xsplit(const float* __restrict__ x,
                          unsigned short* __restrict__ xh,
                          unsigned short* __restrict__ xl) {
    const size_t e = (size_t)blockIdx.x * blockDim.x + threadIdx.x;
    const float4 v = ((const float4*)x)[e];
    const unsigned short ha = f32_to_bf16_rne(v.x);
    const unsigned short hb = f32_to_bf16_rne(v.y);
    const unsigned short hc = f32_to_bf16_rne(v.z);
    const unsigned short hd = f32_to_bf16_rne(v.w);
    const unsigned short la = f32_to_bf16_rne(v.x - bf16_bits_to_f32(ha));
    const unsigned short lb = f32_to_bf16_rne(v.y - bf16_bits_to_f32(hb));
    const unsigned short lc = f32_to_bf16_rne(v.z - bf16_bits_to_f32(hc));
    const unsigned short ld = f32_to_bf16_rne(v.w - bf16_bits_to_f32(hd));
    uint2 hv, lv;
    hv.x = (unsigned)ha | ((unsigned)hb << 16);
    hv.y = (unsigned)hc | ((unsigned)hd << 16);
    lv.x = (unsigned)la | ((unsigned)lb << 16);
    lv.y = (unsigned)lc | ((unsigned)ld << 16);
    ((uint2*)xh)[e] = hv;
    ((uint2*)xl)[e] = lv;
}

// ---------------------------------------------------------------------------
// Prep kernel 4: transpose W [1152,256] -> Wt [256,1152], bf16 hi/lo split
// ---------------------------------------------------------------------------
__global__ void yc_wsplit(const float* __restrict__ w,
                          unsigned short* __restrict__ wh,
                          unsigned short* __restrict__ wl) {
    const int e = blockIdx.x * blockDim.x + threadIdx.x;   // 0..294911
    const int k = e >> 8, o = e & 255;
    const float f  = w[e];
    const unsigned short hb = f32_to_bf16_rne(f);
    const float lo = f - bf16_bits_to_f32(hb);
    wh[(size_t)o * 1152 + k] = hb;
    wl[(size_t)o * 1152 + k] = f32_to_bf16_rne(lo);
}

// ---------------------------------------------------------------------------
// Prep kernel 5: per-output-channel squared kernel norm + zero block
// ---------------------------------------------------------------------------
__global__ void yc_ksq(const float* __restrict__ w, float* __restrict__ ksq,
                       unsigned int* __restrict__ zerobuf) {
    const int o = threadIdx.x;                              // 256 threads
    if (o < 64) zerobuf[o] = 0u;                            // 256 B of zeros
    float acc = 0.0f;
    for (int k = 0; k < 1152; ++k) {
        const float f = w[(size_t)k * 256 + o];
        acc += f * f;
    }
    ksq[o] = acc;
}

// ---------------------------------------------------------------------------
// Main kernel: block tile 128(M) x 128(N), 256 threads = 8 waves (wave32).
// Waves arranged 4(M) x 2(N); each wave owns 2x4 16x16 WMMA tiles.
// K pipeline: 36 steps (9 taps x 4 channel-blocks of 32), ping-pong LDS.
// ---------------------------------------------------------------------------
struct Srcs { const void *ah, *al, *bh, *bl; };

__device__ __forceinline__ Srcs yc_calc_srcs(
        int step, int bb, int h0, int n0, int arow, int ahalf,
        const unsigned short* xh, const unsigned short* xl,
        const unsigned short* wh, const unsigned short* wl, const void* zerop)
{
    const int pos = step >> 2, cb = step & 3;
    const int dh = pos / 3 - 1, dw = pos % 3 - 1;
    const int sh = h0 + (arow >> 6) + dh;
    const int sw = (arow & 63) + dw;
    const bool valid = ((unsigned)sh < 64u) && ((unsigned)sw < 64u);
    const size_t apix = (((size_t)bb * 64 + (valid ? sh : 0)) * 64 + (valid ? sw : 0)) * 128
                        + (size_t)(cb * 32 + ahalf * 16);
    const size_t boff = (size_t)(n0 + arow) * 1152 + pos * 128 + cb * 32 + ahalf * 16;
    Srcs s;
    s.ah = valid ? (const void*)(xh + apix) : zerop;
    s.al = valid ? (const void*)(xl + apix) : zerop;
    s.bh = (const void*)(wh + boff);
    s.bl = (const void*)(wl + boff);
    return s;
}

// issue this thread's 8 async B128 copies for one tile set
// region byte offsets inside one parity buffer: Ah=0, Al=8K, Bh=16K, Bl=24K
__device__ __forceinline__ void yc_stage(const Srcs& s, unsigned ldsBase) {
    async_b128<0>( ldsBase,          s.ah);
    async_b128<16>(ldsBase,          s.ah);
    async_b128<0>( ldsBase + 8192,   s.al);
    async_b128<16>(ldsBase + 8192,   s.al);
    async_b128<0>( ldsBase + 16384,  s.bh);
    async_b128<16>(ldsBase + 16384,  s.bh);
    async_b128<0>( ldsBase + 24576,  s.bl);
    async_b128<16>(ldsBase + 24576,  s.bl);
}

__global__ void __launch_bounds__(256)
yatconv_wmma(const unsigned short* __restrict__ xh,
             const unsigned short* __restrict__ xl,
             const unsigned short* __restrict__ wt_hi,
             const unsigned short* __restrict__ wt_lo,
             const float* __restrict__ ksq,
             const float* __restrict__ bias,
             const float* __restrict__ ps,
             const float* __restrict__ alpha,
             const unsigned int* __restrict__ zerobuf,
             float* __restrict__ out)
{
    // ping-pong tile store: [parity][region Ah|Al|Bh|Bl][128 rows][32 bf16]
    __shared__ __align__(16) unsigned short sT[2 * 4 * 128 * 32];   // 64 KiB
    __shared__ float sPS[128];

    const int t      = threadIdx.x;
    const int lane   = t & 31;
    const int l15    = lane & 15;
    const int half   = lane >> 4;
    const int wid    = t >> 5;
    const int wave_m = wid >> 1;          // 0..3
    const int wave_n = wid & 1;           // 0..1

    const int m0 = blockIdx.x * 128;      // pixel base
    const int n0 = blockIdx.y * 128;      // out-channel base
    const int bb = m0 >> 12;
    const int h0 = (m0 >> 6) & 63;

    if (t < 128) sPS[t] = ps[m0 + t];

    const f32x8 vzero = {};
    f32x8 acc[2][4];
#pragma unroll
    for (int mt = 0; mt < 2; ++mt)
#pragma unroll
        for (int nt = 0; nt < 4; ++nt) acc[mt][nt] = vzero;

    const int arow  = t >> 1;             // 0..127 : tile row staged by this thread
    const int ahalf = t & 1;              // which 16-element half of the 32-wide K slice

    const unsigned ldsStage = lds_off(sT) + (unsigned)(arow * 64 + ahalf * 32);
    const void* zerop = (const void*)zerobuf;

    // ---- pipeline prologue: stage step 0 into parity 0 --------------------
    {
        Srcs s0 = yc_calc_srcs(0, bb, h0, n0, arow, ahalf, xh, xl, wt_hi, wt_lo, zerop);
        yc_stage(s0, ldsStage);
    }

    for (int step = 0; step < 36; ++step) {
        const int par = step & 1;

        // ---- stage step+1 into the other parity, then wait for step's tiles.
        // ASYNCcnt retires in order, so <=8 outstanding means step's 8 landed.
        if (step < 35) {
            Srcs sn = yc_calc_srcs(step + 1, bb, h0, n0, arow, ahalf,
                                   xh, xl, wt_hi, wt_lo, zerop);
            yc_stage(sn, ldsStage + (unsigned)(((step + 1) & 1) * 32768));
            // prefetch weights two steps ahead (global_prefetch_b8)
            if (step < 34)
                __builtin_prefetch((const char*)sn.bh + 64, 0, 1);
            wait_async<8>();
        } else {
            wait_async<0>();
        }
        __syncthreads();                   // all waves' step tiles visible

        // ---- per-lane fragments matching ISA VGPR layouts ------------------
        const unsigned short* buf = sT + par * 16384;
        const uint4* bAh = (const uint4*)(buf);
        const uint4* bAl = (const uint4*)(buf + 4096);
        const uint4* bBh = (const uint4*)(buf + 8192);
        const uint4* bBl = (const uint4*)(buf + 12288);

        bf16x16 aH[2], aL[2], bH[4], bL[4];
#pragma unroll
        for (int mt = 0; mt < 2; ++mt) {
            const int r = (wave_m * 2 + mt) * 16 + l15;
            // A 16-bit 16x32: lane half selects K {0..7,16..23} vs {8..15,24..31}
            aH[mt] = make_frag(bAh[r * 4 + half], bAh[r * 4 + 2 + half]);
            aL[mt] = make_frag(bAl[r * 4 + half], bAl[r * 4 + 2 + half]);
        }
#pragma unroll
        for (int nt = 0; nt < 4; ++nt) {
            const int r = (wave_n * 4 + nt) * 16 + l15;
            // B 32x16: lane half selects K {0..7,16..23} vs {8..15,24..31} (matches A)
            bH[nt] = make_frag(bBh[r * 4 + half], bBh[r * 4 + 2 + half]);
            bL[nt] = make_frag(bBl[r * 4 + half], bBl[r * 4 + 2 + half]);
        }

        // ---- 3-WMMA error-compensated fp32 product -------------------------
#pragma unroll
        for (int mt = 0; mt < 2; ++mt) {
#pragma unroll
            for (int nt = 0; nt < 4; ++nt) {
                acc[mt][nt] = __builtin_amdgcn_wmma_f32_16x16x32_bf16(
                    false, aH[mt], false, bH[nt], (short)0, acc[mt][nt], false, false);
                acc[mt][nt] = __builtin_amdgcn_wmma_f32_16x16x32_bf16(
                    false, aH[mt], false, bL[nt], (short)0, acc[mt][nt], false, false);
                acc[mt][nt] = __builtin_amdgcn_wmma_f32_16x16x32_bf16(
                    false, aL[mt], false, bH[nt], (short)0, acc[mt][nt], false, false);
            }
        }

        __syncthreads();   // all waves done reading parity `par` before it is
                           // restaged at step+2 (WAR protection for ping-pong)
    }

    // ---- epilogue: y = dot^2/(||x||^2 - 2dot + ||W||^2 + eps) + bias, * scale
    const float av    = alpha[0];
    const float base  = 16.0f / log1pf(256.0f);   // sqrt(256)/log1p(256)
    const float scale = powf(base, av);

#pragma unroll
    for (int mt = 0; mt < 2; ++mt) {
#pragma unroll
        for (int nt = 0; nt < 4; ++nt) {
            const int n  = n0 + (wave_n * 4 + nt) * 16 + l15;
            const float kq = ksq[n];
            const float bs = bias[n];
#pragma unroll
            for (int j = 0; j < 8; ++j) {
                const int rloc = (wave_m * 2 + mt) * 16 + half * 8 + j;
                const float dot  = acc[mt][nt][j];
                const float dist = sPS[rloc] - 2.0f * dot + kq;
                const float y    = (dot * dot) / (dist + YC_EPS) + bs;
                out[(size_t)(m0 + rloc) * 256 + n] = y * scale;
            }
        }
    }
}

// ---------------------------------------------------------------------------
// Launcher
// ---------------------------------------------------------------------------
extern "C" void kernel_launch(void* const* d_in, const int* in_sizes, int n_in,
                              void* d_out, int out_size, void* d_ws, size_t ws_size,
                              hipStream_t stream) {
    (void)in_sizes; (void)n_in; (void)out_size; (void)ws_size;

    const float* x      = (const float*)d_in[0];   // [16,64,64,128]
    const float* kernel = (const float*)d_in[1];   // [3,3,128,256]
    const float* bias   = (const float*)d_in[2];   // [256]
    const float* alpha  = (const float*)d_in[3];   // [1]
    float* out = (float*)d_out;                    // [16,64,64,256]

    // workspace layout (aligned sections)
    char* ws = (char*)d_ws;
    float*          s    = (float*)(ws);                          //  65536 f  (256 KiB)
    float*          ps   = (float*)(ws + 262144);                 //  65536 f  (256 KiB)
    unsigned short* wh   = (unsigned short*)(ws + 524288);        // 294912 u16 (576 KiB)
    unsigned short* wl   = (unsigned short*)(ws + 1114112);       // 294912 u16 (576 KiB)
    float*          ksq  = (float*)(ws + 1703936);                //    256 f
    unsigned int*   zb   = (unsigned int*)(ws + 1704960);         //    256 B zeros
    unsigned short* xh   = (unsigned short*)(ws + 1705216);       // 16 MiB
    unsigned short* xl   = (unsigned short*)(ws + 1705216 + 16777216);
    // total ~33.6 MiB

    yc_pix_sumsq<<<256,  256, 0, stream>>>(x, s);
    yc_boxsum  <<<256,  256, 0, stream>>>(s, ps);
    yc_xsplit  <<<8192, 256, 0, stream>>>(x, xh, xl);
    yc_wsplit  <<<1152, 256, 0, stream>>>(kernel, wh, wl);
    yc_ksq     <<<1,    256, 0, stream>>>(kernel, ksq, zb);

    dim3 grid(512, 2);   // 65536/128 M-tiles x 256/128 N-tiles
    yatconv_wmma<<<grid, 256, 0, stream>>>(xh, xl, wh, wl, ksq, bias, ps, alpha, zb, out);
}
